// LSTMModel_18004502905292
// MI455X (gfx1250) — matrix operand order repaired and checked by reading, not compiled, not run on hardware
//
#include <hip/hip_runtime.h>
#include <hip/hip_bf16.h>

// MI455X / gfx1250, wave32. WMMA bf16 (16x16x32, f32 accum) for all GEMM work.
// Recurrent kernels double-buffer xw[t] tiles into LDS via async global->LDS.

#define DEVINL static __device__ __forceinline__

typedef __attribute__((ext_vector_type(16))) __bf16         v16bf;
typedef __attribute__((ext_vector_type(16))) unsigned short v16u;
typedef __attribute__((ext_vector_type(8)))  unsigned short v8u;
typedef __attribute__((ext_vector_type(8)))  float          v8f;
typedef __attribute__((ext_vector_type(4)))  float          v4f;

union AFrag {
  v16u v;
  v8u h[2];
  unsigned short s[16];
};

DEVINL unsigned short f2bf(float f) {
  union { float f; unsigned int u; } cv;
  cv.f = f;
  unsigned int r = cv.u + 0x7FFFu + ((cv.u >> 16) & 1u);  // RNE
  return (unsigned short)(r >> 16);
}

DEVINL v8f wmma_bf16(v16u a, v16u b, v8f c) {
  return __builtin_amdgcn_wmma_f32_16x16x32_bf16(
      false, __builtin_bit_cast(v16bf, a),
      false, __builtin_bit_cast(v16bf, b),
      (short)0, c, false, false);
}

DEVINL float sigmoidf_(float x) { return 1.0f / (1.0f + __expf(-x)); }

// Async-copy one contiguous 16 KB wave tile (16 batch rows x 256 gates, f32)
// from global memory into LDS: 32 x global_load_async_to_lds_b128
// (each moves 32 lanes x 16 B = 512 B). Tracked by ASYNCcnt.
DEVINL void stage_async(const float* __restrict__ xw, int t, int bbase_u,
                        unsigned lds_base, int lane) {
  const char* sbase = (const char*)(xw + ((size_t)t * 1024 + bbase_u) * 256);
#pragma unroll
  for (int i = 0; i < 32; ++i) {
    const unsigned off = (unsigned)((i * 32 + lane) * 16);
    asm volatile("global_load_async_to_lds_b128 %0, %1, %2"
                 :
                 : "v"(lds_base + off), "v"(off), "s"(sbase)
                 : "memory");
  }
}

// ---------------------------------------------------------------------------
// Kernel 1: xw0[t,b,:] = x[b,t,:] @ Wih0^T + (bih0+bhh0)
// M = B*T = 262144, K = 256, N = 256. Grid: (M/128, 2 N-halves).
// 8 waves/WG, each wave: 16 rows x 128 cols, K swept in 8 WMMA steps.
// Wih0 half (128x256) converted to bf16 in LDS (64 KB).
// ---------------------------------------------------------------------------
__global__ __launch_bounds__(256) void k_gemm_in0(
    const float* __restrict__ x, const float* __restrict__ W,
    const float* __restrict__ b0, const float* __restrict__ b1,
    float* __restrict__ xw) {
  extern __shared__ unsigned short ldsW[];  // [128][256] bf16
  const int tid = threadIdx.x;
  const int nbase0 = blockIdx.y * 128;
  for (int i = tid; i < 128 * 256; i += 256) ldsW[i] = f2bf(W[nbase0 * 256 + i]);
  __syncthreads();

  const int lane = tid & 31;
  const int wave = tid >> 5;
  const int hi = lane >> 4;    // lane half selects K sub-chunk
  const int ncol = lane & 15;  // column within 16-wide N tile
  const int mbase = blockIdx.x * 128 + wave * 16;

  v8f acc[8];
#pragma unroll
  for (int nt = 0; nt < 8; ++nt) {
    const int gidx = nbase0 + nt * 16 + ncol;
    const float bv = b0[gidx] + b1[gidx];
#pragma unroll
    for (int r = 0; r < 8; ++r) acc[nt][r] = bv;
  }

  const float* xr = x + (size_t)(mbase + (lane & 15)) * 256;
#pragma unroll
  for (int kt = 0; kt < 8; ++kt) {
    const int kb = kt * 32 + hi * 16;
    v4f c0 = *(const v4f*)(xr + kb);
    v4f c1 = *(const v4f*)(xr + kb + 4);
    v4f c2 = *(const v4f*)(xr + kb + 8);
    v4f c3 = *(const v4f*)(xr + kb + 12);
    AFrag a;
#pragma unroll
    for (int e = 0; e < 4; ++e) {
      a.s[e] = f2bf(c0[e]);
      a.s[4 + e] = f2bf(c1[e]);
      a.s[8 + e] = f2bf(c2[e]);
      a.s[12 + e] = f2bf(c3[e]);
    }
#pragma unroll
    for (int nt = 0; nt < 8; ++nt) {
      const unsigned short* wr = ldsW + (nt * 16 + ncol) * 256 + kt * 32 + hi * 16;
      AFrag bf;
      bf.h[0] = *(const v8u*)(wr);
      bf.h[1] = *(const v8u*)(wr + 8);
      acc[nt] = wmma_bf16(a.v, bf.v, acc[nt]);
    }
  }

  // scatter to [T,B,4H] layout (rows m = b*T + t)
#pragma unroll
  for (int nt = 0; nt < 8; ++nt) {
#pragma unroll
    for (int r = 0; r < 8; ++r) {
      const int m = mbase + r + hi * 8;
      const int bb = m >> 8;   // T = 256
      const int tt = m & 255;
      xw[((size_t)tt * 1024 + bb) * 256 + nbase0 + nt * 16 + ncol] = acc[nt][r];
    }
  }
}

// ---------------------------------------------------------------------------
// Kernels 2/3: persistent LSTM recurrence. 2 waves/WG, each wave owns 16
// batch rows; grid = 1024/32 = 32 WGs (spreads across WGPs). Whh (and, when
// FUSE, the next layer's Wih + fused bias) live in LDS as bf16. h round-trips
// through a wave-private LDS scratch to re-shape D-layout -> A-fragments.
// xw[t] wave tiles are double-buffered into LDS with async global->LDS copies
// overlapping the previous step's WMMAs (s_wait_asynccnt <= 32 to keep the
// t+1 prefetch in flight while t's buffer is guaranteed complete).
// FUSE: also emits xw_out[t] = h[t] @ Wp^T + bias (layer-1 input projection).
// ---------------------------------------------------------------------------
template <bool FUSE>
__global__ __launch_bounds__(64) void k_lstm(
    const float* __restrict__ xw, const float* __restrict__ Whh,
    const float* __restrict__ Wp, const float* __restrict__ bpA,
    const float* __restrict__ bpB, float* __restrict__ xw_out,
    float* __restrict__ h_last) {
  extern __shared__ unsigned char smem[];
  unsigned short* ldsWhh = (unsigned short*)smem;         // [256][64] bf16
  unsigned short* ldsWp = ldsWhh + 256 * 64;              // [256][64] bf16 (FUSE)
  float* ldsBp = (float*)(ldsWp + 256 * 64);              // [256] f32 (FUSE)
  unsigned short* hscr = (unsigned short*)(ldsBp + 256);  // per-wave [16][64]
  float* xs = (float*)(hscr + 2 * 16 * 64);               // [2][2][16*256] f32

  const int tid = threadIdx.x;
  for (int i = tid; i < 256 * 64; i += 64) {
    ldsWhh[i] = f2bf(Whh[i]);
    if (FUSE) ldsWp[i] = f2bf(Wp[i]);
  }
  if (FUSE)
    for (int i = tid; i < 256; i += 64) ldsBp[i] = bpA[i] + bpB[i];

  const int lane = tid & 31;
  const int wave = tid >> 5;
  unsigned short* hs = hscr + wave * (16 * 64);
  for (int i = lane; i < 16 * 64; i += 32) hs[i] = 0;
  __syncthreads();

  const int hi = lane >> 4;
  const int ncol = lane & 15;
  const int mrow = lane & 15;
  const int bbase = blockIdx.x * 32 + wave * 16;
  const int bbase_u = __builtin_amdgcn_readfirstlane(bbase);

  float* myxs[2] = {xs + (0 * 2 + wave) * (16 * 256),
                    xs + (1 * 2 + wave) * (16 * 256)};
  const unsigned lds_x[2] = {(unsigned)(uintptr_t)(void*)myxs[0],
                             (unsigned)(uintptr_t)(void*)myxs[1]};

  v8f cst[4];  // cell state, D-layout, H = 4 tiles of 16
#pragma unroll
  for (int j = 0; j < 4; ++j)
#pragma unroll
    for (int r = 0; r < 8; ++r) cst[j][r] = 0.f;

  AFrag ha[2];  // h as A-fragments (K = 64 -> 2 WMMA k-steps)
#pragma unroll
  for (int kt = 0; kt < 2; ++kt)
#pragma unroll
    for (int e = 0; e < 16; ++e) ha[kt].s[e] = 0;

  // prime the pipeline: stage xw[0] into buffer 0
  stage_async(xw, 0, bbase_u, lds_x[0], lane);

  for (int t = 0; t < 256; ++t) {
    const int p = t & 1;
    if (t + 1 < 256) {
      stage_async(xw, t + 1, bbase_u, lds_x[p ^ 1], lane);
      // oldest 32 async ops (this step's buffer) retired; t+1 stays in flight
      asm volatile("s_wait_asynccnt 0x20" ::: "memory");
    } else {
      asm volatile("s_wait_asynccnt 0x0" ::: "memory");
    }
    const float* xrow = myxs[p];

    // gates = xw[t] + h @ Whh^T  (f32 accumulate)
    v8f g[16];
#pragma unroll
    for (int nt = 0; nt < 16; ++nt) {
#pragma unroll
      for (int r = 0; r < 8; ++r) {
        g[nt][r] = xrow[(r + hi * 8) * 256 + nt * 16 + ncol];
      }
    }
#pragma unroll
    for (int kt = 0; kt < 2; ++kt) {
#pragma unroll
      for (int nt = 0; nt < 16; ++nt) {
        const unsigned short* wr =
            ldsWhh + (nt * 16 + ncol) * 64 + kt * 32 + hi * 16;
        AFrag bf;
        bf.h[0] = *(const v8u*)(wr);
        bf.h[1] = *(const v8u*)(wr + 8);
        g[nt] = wmma_bf16(ha[kt].v, bf.v, g[nt]);
      }
    }

    // LSTM cell elementwise; write h (bf16) to wave-private LDS scratch
#pragma unroll
    for (int j = 0; j < 4; ++j) {
#pragma unroll
      for (int r = 0; r < 8; ++r) {
        const float iv = sigmoidf_(g[j][r]);
        const float fv = sigmoidf_(g[j + 4][r]);
        const float gv = tanhf(g[j + 8][r]);
        const float ov = sigmoidf_(g[j + 12][r]);
        const float cv = fv * cst[j][r] + iv * gv;
        cst[j][r] = cv;
        const float hv = ov * tanhf(cv);
        hs[(r + hi * 8) * 64 + j * 16 + ncol] = f2bf(hv);
        if (!FUSE && t == 255) {
          h_last[(size_t)(bbase + r + hi * 8) * 64 + j * 16 + ncol] = hv;
        }
      }
    }

    // reload h as A-fragments (used by fused projection AND next iteration)
#pragma unroll
    for (int kt = 0; kt < 2; ++kt) {
      const unsigned short* hr = hs + mrow * 64 + kt * 32 + hi * 16;
      ha[kt].h[0] = *(const v8u*)(hr);
      ha[kt].h[1] = *(const v8u*)(hr + 8);
    }

    if (FUSE) {  // xw_out[t] = bias1 + h @ Wp^T  (layer-1 input projection)
      float* orow = xw_out + (size_t)t * (1024 * 256);
      v8f pacc[16];
#pragma unroll
      for (int nt = 0; nt < 16; ++nt) {
        const float bv = ldsBp[nt * 16 + ncol];
#pragma unroll
        for (int r = 0; r < 8; ++r) pacc[nt][r] = bv;
      }
#pragma unroll
      for (int kt = 0; kt < 2; ++kt) {
#pragma unroll
        for (int nt = 0; nt < 16; ++nt) {
          const unsigned short* wr =
              ldsWp + (nt * 16 + ncol) * 64 + kt * 32 + hi * 16;
          AFrag bf;
          bf.h[0] = *(const v8u*)(wr);
          bf.h[1] = *(const v8u*)(wr + 8);
          pacc[nt] = wmma_bf16(ha[kt].v, bf.v, pacc[nt]);
        }
      }
#pragma unroll
      for (int nt = 0; nt < 16; ++nt) {
#pragma unroll
        for (int r = 0; r < 8; ++r) {
          orow[(size_t)(bbase + r + hi * 8) * 256 + nt * 16 + ncol] =
              pacc[nt][r];
        }
      }
    }
  }
}

// ---------------------------------------------------------------------------
// Kernel 4: BatchNorm over batch (biased var) -> normalized h
// ---------------------------------------------------------------------------
__global__ __launch_bounds__(64) void k_bn(const float* __restrict__ h,
                                           const float* __restrict__ gamma,
                                           const float* __restrict__ beta,
                                           float* __restrict__ nh) {
  const int j = threadIdx.x;  // feature 0..63
  float s = 0.f, s2 = 0.f;
  for (int b = 0; b < 1024; ++b) {
    const float v = h[b * 64 + j];
    s += v;
    s2 += v * v;
  }
  const float mean = s * (1.f / 1024.f);
  const float var = s2 * (1.f / 1024.f) - mean * mean;
  const float sc = gamma[j] * rsqrtf(var + 1e-5f);
  const float sh = beta[j] - mean * sc;
  for (int b = 0; b < 1024; ++b) nh[b * 64 + j] = h[b * 64 + j] * sc + sh;
}

// ---------------------------------------------------------------------------
// Kernel 5: MLP trunk + 3 heads (one thread per batch row; weights in LDS)
// ---------------------------------------------------------------------------
__global__ __launch_bounds__(256) void k_mlp(
    const float* __restrict__ nh, const float* __restrict__ W1,
    const float* __restrict__ b1, const float* __restrict__ W2,
    const float* __restrict__ b2, const float* __restrict__ Wmc,
    const float* __restrict__ bmc, const float* __restrict__ Wco,
    const float* __restrict__ bco, const float* __restrict__ Wip,
    const float* __restrict__ bip, float* __restrict__ out) {
  __shared__ float sW1[64 * 64];
  __shared__ float sW2[32 * 64];
  __shared__ float sWh[7 * 32];
  const int tid = threadIdx.x;
  for (int i = tid; i < 64 * 64; i += 256) sW1[i] = W1[i];
  for (int i = tid; i < 32 * 64; i += 256) sW2[i] = W2[i];
  for (int i = tid; i < 3 * 32; i += 256) sWh[i] = Wmc[i];
  for (int i = tid; i < 2 * 32; i += 256) sWh[96 + i] = Wco[i];
  for (int i = tid; i < 2 * 32; i += 256) sWh[160 + i] = Wip[i];
  __syncthreads();

  const int row = blockIdx.x * 256 + tid;
  float xin[64];
  for (int i = 0; i < 64; ++i) xin[i] = nh[(size_t)row * 64 + i];
  float h1[64];
  for (int o = 0; o < 64; ++o) {
    float a = b1[o];
    for (int i = 0; i < 64; ++i) a = fmaf(sW1[o * 64 + i], xin[i], a);
    h1[o] = fmaxf(a, 0.f);
  }
  float h2[32];
  for (int o = 0; o < 32; ++o) {
    float a = b2[o];
    for (int i = 0; i < 64; ++i) a = fmaf(sW2[o * 64 + i], h1[i], a);
    h2[o] = fmaxf(a, 0.f);
  }
  for (int k = 0; k < 3; ++k) {
    float a = bmc[k];
    for (int i = 0; i < 32; ++i) a = fmaf(sWh[k * 32 + i], h2[i], a);
    out[(size_t)row * 3 + k] = a;
  }
  for (int k = 0; k < 2; ++k) {
    float a = bco[k];
    for (int i = 0; i < 32; ++i) a = fmaf(sWh[96 + k * 32 + i], h2[i], a);
    out[3072 + (size_t)row * 2 + k] = a;
  }
  for (int k = 0; k < 2; ++k) {
    float a = bip[k];
    for (int i = 0; i < 32; ++i) a = fmaf(sWh[160 + k * 32 + i], h2[i], a);
    out[5120 + (size_t)row * 2 + k] = a;
  }
}

// ---------------------------------------------------------------------------
extern "C" void kernel_launch(void* const* d_in, const int* in_sizes, int n_in,
                              void* d_out, int out_size, void* d_ws,
                              size_t ws_size, hipStream_t stream) {
  const float* x = (const float*)d_in[0];
  const float* Wih0 = (const float*)d_in[1];
  const float* Whh0 = (const float*)d_in[2];
  const float* bih0 = (const float*)d_in[3];
  const float* bhh0 = (const float*)d_in[4];
  const float* Wih1 = (const float*)d_in[5];
  const float* Whh1 = (const float*)d_in[6];
  const float* bih1 = (const float*)d_in[7];
  const float* bhh1 = (const float*)d_in[8];
  const float* gamma = (const float*)d_in[9];
  const float* beta = (const float*)d_in[10];
  const float* Wfc1 = (const float*)d_in[11];
  const float* bfc1 = (const float*)d_in[12];
  const float* Wfc2 = (const float*)d_in[13];
  const float* bfc2 = (const float*)d_in[14];
  const float* Wmc = (const float*)d_in[15];
  const float* bmc = (const float*)d_in[16];
  const float* Wco = (const float*)d_in[17];
  const float* bco = (const float*)d_in[18];
  const float* Wip = (const float*)d_in[19];
  const float* bip = (const float*)d_in[20];

  float* xw0 = (float*)d_ws;                      // [256,1024,256] f32
  float* xw1 = xw0 + (size_t)256 * 1024 * 256;    // [256,1024,256] f32
  float* hlast = xw1 + (size_t)256 * 1024 * 256;  // [1024,64]
  float* nh = hlast + 1024 * 64;                  // [1024,64]

  const size_t lds1 = (size_t)128 * 256 * sizeof(unsigned short);  // 64 KB
  const size_t lds2 = (size_t)2 * 256 * 64 * sizeof(unsigned short) +
                      256 * sizeof(float) +
                      (size_t)2 * 16 * 64 * sizeof(unsigned short) +
                      (size_t)2 * 2 * 16 * 256 * sizeof(float);  // ~133 KB

  (void)hipFuncSetAttribute(reinterpret_cast<const void*>(k_lstm<true>),
                            hipFuncAttributeMaxDynamicSharedMemorySize,
                            (int)lds2);
  (void)hipFuncSetAttribute(reinterpret_cast<const void*>(k_lstm<false>),
                            hipFuncAttributeMaxDynamicSharedMemorySize,
                            (int)lds2);

  k_gemm_in0<<<dim3(2048, 2), 256, lds1, stream>>>(x, Wih0, bih0, bhh0, xw0);
  k_lstm<true><<<32, 64, lds2, stream>>>(xw0, Whh0, Wih1, bih1, bhh1, xw1,
                                         nullptr);
  k_lstm<false><<<32, 64, lds2, stream>>>(xw1, Whh1, nullptr, nullptr, nullptr,
                                          nullptr, hlast);
  k_bn<<<1, 64, 0, stream>>>(hlast, gamma, beta, nh);
  k_mlp<<<4, 256, 0, stream>>>(nh, Wfc1, bfc1, Wfc2, bfc2, Wmc, bmc, Wco, bco,
                               Wip, bip, (float*)d_out);
}
